// PointNet2Sem_seg_MSG_29076928594037
// MI455X (gfx1250) — compile-verified, matrix operand order checked by reading, not run on hardware
//
#include <hip/hip_runtime.h>
#include <math.h>

// ---------------------------------------------------------------------------
// PointNet++ MSG semantic segmentation forward, MI455X (gfx1250, wave32).
// GEMMs: operands pre-packed to fragment-native f16 tiles once per layer,
// then a minimal loop of coalesced global_load_b128 -> v_wmma_f32_16x16x32_f16.
// ---------------------------------------------------------------------------

typedef __attribute__((ext_vector_type(16))) _Float16 v16h;
typedef __attribute__((ext_vector_type(8)))  float    v8f;

#define BN_EPS 1e-5f

// ---------------------------------------------------------------------------
__global__ void extract_xyz_kernel(const float* __restrict__ x, float* __restrict__ xyz,
                                   int total /* B*N */) {
    int i = blockIdx.x * blockDim.x + threadIdx.x;
    if (i >= total) return;
    xyz[i * 3 + 0] = x[i * 9 + 0];
    xyz[i * 3 + 1] = x[i * 9 + 1];
    xyz[i * 3 + 2] = x[i * 9 + 2];
}

// ---------------------------------------------------------------------------
// Farthest point sampling: one block per batch, iterative argmax with an LDS
// tree reduction (tie-break to lowest index to match jnp.argmax).
__global__ void fps_kernel(const float* __restrict__ xyz, float* __restrict__ dist,
                           int* __restrict__ out, int N, int np) {
    const int b = blockIdx.x;
    const int t = threadIdx.x;
    __shared__ float sv[256];
    __shared__ int   si[256];
    __shared__ int   sfar;
    for (int n = t; n < N; n += 256) dist[b * N + n] = 1e10f;
    if (t == 0) sfar = 0;
    __syncthreads();
    for (int it = 0; it < np; ++it) {
        const int far = sfar;
        if (t == 0) out[b * np + it] = far;
        const float cx = xyz[(b * N + far) * 3 + 0];
        const float cy = xyz[(b * N + far) * 3 + 1];
        const float cz = xyz[(b * N + far) * 3 + 2];
        float best = -1.0f; int bi = 0;
        for (int n = t; n < N; n += 256) {
            const float dx = xyz[(b * N + n) * 3 + 0] - cx;
            const float dy = xyz[(b * N + n) * 3 + 1] - cy;
            const float dz = xyz[(b * N + n) * 3 + 2] - cz;
            const float d  = dx * dx + dy * dy + dz * dz;
            float dn = fminf(dist[b * N + n], d);
            dist[b * N + n] = dn;
            if (dn > best) { best = dn; bi = n; }
        }
        sv[t] = best; si[t] = bi;
        __syncthreads();
        for (int s = 128; s > 0; s >>= 1) {
            if (t < s) {
                if (sv[t + s] > sv[t] || (sv[t + s] == sv[t] && si[t + s] < si[t])) {
                    sv[t] = sv[t + s]; si[t] = si[t + s];
                }
            }
            __syncthreads();
        }
        if (t == 0) sfar = si[0];
        __syncthreads();
    }
}

// ---------------------------------------------------------------------------
__global__ void gather_xyz_kernel(const float* __restrict__ xyz, const int* __restrict__ idx,
                                  float* __restrict__ out, int N, int total /* B*S */, int S) {
    int i = blockIdx.x * blockDim.x + threadIdx.x;
    if (i >= total) return;
    const int b = i / S;
    const int n = idx[i];
    out[i * 3 + 0] = xyz[(b * N + n) * 3 + 0];
    out[i * 3 + 1] = xyz[(b * N + n) * 3 + 1];
    out[i * 3 + 2] = xyz[(b * N + n) * 3 + 2];
}

// ---------------------------------------------------------------------------
__global__ void ball_group_kernel(const float* __restrict__ xyz, const float* __restrict__ new_xyz,
                                  int* __restrict__ gidx, int N, int S, int total /* B*S */,
                                  float r2, int ns) {
    int i = blockIdx.x * blockDim.x + threadIdx.x;
    if (i >= total) return;
    const int b = i / S;
    const float cx = new_xyz[i * 3 + 0];
    const float cy = new_xyz[i * 3 + 1];
    const float cz = new_xyz[i * 3 + 2];
    int cnt = 0, first = -1;
    for (int n = 0; n < N && cnt < ns; ++n) {
        const float dx = xyz[(b * N + n) * 3 + 0] - cx;
        const float dy = xyz[(b * N + n) * 3 + 1] - cy;
        const float dz = xyz[(b * N + n) * 3 + 2] - cz;
        const float d  = dx * dx + dy * dy + dz * dz;
        if (d <= r2) {
            if (first < 0) first = n;
            gidx[i * ns + cnt] = n;
            ++cnt;
        }
    }
    if (first < 0) first = 0;
    for (; cnt < ns; ++cnt) gidx[i * ns + cnt] = first;
}

// ---------------------------------------------------------------------------
__global__ void group_concat_kernel(const float* __restrict__ xyz, const float* __restrict__ feat,
                                    const float* __restrict__ new_xyz, const int* __restrict__ gidx,
                                    float* __restrict__ out, int N, int S, int ns, int C,
                                    int total /* B*S*ns */) {
    int i = blockIdx.x * blockDim.x + threadIdx.x;
    if (i >= total) return;
    const int b   = i / (S * ns);
    const int rem = i % (S * ns);
    const int s   = rem / ns;
    const int n   = gidx[i];
    float* o = out + (size_t)i * (C + 3);
    const float* f = feat + (size_t)(b * N + n) * C;
    for (int c = 0; c < C; ++c) o[c] = f[c];
    o[C + 0] = xyz[(b * N + n) * 3 + 0] - new_xyz[(b * S + s) * 3 + 0];
    o[C + 1] = xyz[(b * N + n) * 3 + 1] - new_xyz[(b * S + s) * 3 + 1];
    o[C + 2] = xyz[(b * N + n) * 3 + 2] - new_xyz[(b * S + s) * 3 + 2];
}

// ---------------------------------------------------------------------------
// Pack X[M,K] (f32) into fragment-native f16 tiles:
//   Xp[rowTile][kTile][lane 0..31][16 halves], K zero-padded to 32*nkt.
// Per the CDNA5 16-bit A layout, lane (hi,l16) holds row rowTile*16+l16,
// k-chunks [kt*32 + hi*8 .. +8) and [kt*32 + 16 + hi*8 .. +8).
// One thread per contiguous 16-B output chunk (coalesced stores).
__global__ void pack_a_kernel(const float* __restrict__ X, _Float16* __restrict__ Xp,
                              int K, int nkt, int totalChunks) {
    int t = blockIdx.x * blockDim.x + threadIdx.x;
    if (t >= totalChunks) return;
    const int c    = t & 1;                 // chunk within lane entry
    const int lane = (t >> 1) & 31;
    const int kt   = (t >> 6) % nkt;
    const int rt   = (t >> 6) / nkt;
    const int row  = rt * 16 + (lane & 15);
    const int kb   = kt * 32 + c * 16 + (lane >> 4) * 8;
    _Float16* o = Xp + (size_t)t * 8;
    const float* xr = X + (size_t)row * K;
#pragma unroll
    for (int j = 0; j < 8; ++j) {
        const int k = kb + j;
        float v = xr[k < K ? k : K - 1];    // clamped: always valid
        o[j] = (_Float16)(k < K ? v : 0.0f);
    }
}

// Pack W[K,N] (f32) into B-fragment tiles: Wp[colTile][kTile][lane][16 halves].
__global__ void pack_b_kernel(const float* __restrict__ W, _Float16* __restrict__ Wp,
                              int K, int N, int nkt, int totalChunks) {
    int t = blockIdx.x * blockDim.x + threadIdx.x;
    if (t >= totalChunks) return;
    const int c    = t & 1;
    const int lane = (t >> 1) & 31;
    const int kt   = (t >> 6) % nkt;
    const int ct   = (t >> 6) / nkt;
    const int col  = ct * 16 + (lane & 15);
    const int kb   = kt * 32 + c * 16 + (lane >> 4) * 8;
    const int cc   = col < N ? col : N - 1;
    _Float16* o = Wp + (size_t)t * 8;
#pragma unroll
    for (int j = 0; j < 8; ++j) {
        const int k = kb + j;
        float v = W[(size_t)(k < K ? k : K - 1) * N + cc];
        o[j] = (_Float16)((k < K && col < N) ? v : 0.0f);
    }
}

// ---------------------------------------------------------------------------
// Packed WMMA GEMM: Y[M,N] = unpack(Xp) @ unpack(Wp) + bias.
// Block = 128 threads (4 waves), each wave owns one 16x16 output tile.
// Inner loop: 2x 32-B coalesced loads + prefetch + v_wmma. No branches,
// EXEC all-ones throughout. Requires M % 64 == 0 (true at every call site).
__global__ __launch_bounds__(128) void gemm_packed_wmma(const _Float16* __restrict__ Xp,
                                                        const _Float16* __restrict__ Wp,
                                                        const float* __restrict__ bias,
                                                        float* __restrict__ Y,
                                                        int M, int N, int nkt) {
    const int lane    = threadIdx.x & 31;
    const int wave    = threadIdx.x >> 5;
    const int rowTile = blockIdx.y * 4 + wave;
    const int colTile = blockIdx.x;

    const _Float16* pa = Xp + ((size_t)rowTile * nkt * 32 + lane) * 16;
    const _Float16* pb = Wp + ((size_t)colTile * nkt * 32 + lane) * 16;

    v8f acc = {};
    for (int kt = 0; kt < nkt; ++kt) {
        __builtin_prefetch(pa + 512, 0, 1);          // next A k-tile
        const v16h a = *(const v16h*)pa;
        const v16h b = *(const v16h*)pb;
        pa += 512;                                   // 32 lanes * 16 halves
        pb += 512;
        acc = __builtin_amdgcn_wmma_f32_16x16x32_f16(false, a, false, b,
                                                     (short)0, acc, false, false);
    }

    const int c = colTile * 16 + (lane & 15);
    if (c < N) {
        const float bb = bias[c];
        const int hi = lane >> 4;
#pragma unroll
        for (int v = 0; v < 8; ++v) {
            const int r = rowTile * 16 + v + (hi << 3);
            Y[(size_t)r * N + c] = acc[v] + bb;
        }
    }
}

// ---------------------------------------------------------------------------
__global__ void bn_stats_kernel(const float* __restrict__ Y, float* __restrict__ stats,
                                int M, int N) {
    const int c = blockIdx.x * blockDim.x + threadIdx.x;
    if (c >= N) return;
    const int r0 = blockIdx.y * 4096;
    const int r1 = min(r0 + 4096, M);
    float s = 0.0f, s2 = 0.0f;
    for (int r = r0; r < r1; ++r) {
        const float v = Y[(size_t)r * N + c];
        s += v; s2 += v * v;
    }
    atomicAdd(&stats[c], s);
    atomicAdd(&stats[N + c], s2);
}

__global__ void bn_relu_kernel(float* __restrict__ Y, const float* __restrict__ stats,
                               const float* __restrict__ gamma, const float* __restrict__ beta,
                               int M, int N, long long total) {
    long long i = (long long)blockIdx.x * blockDim.x + threadIdx.x;
    if (i >= total) return;
    const int c = (int)(i % N);
    const float mean = stats[c] / (float)M;
    const float var  = stats[N + c] / (float)M - mean * mean;
    float v = (Y[i] - mean) * rsqrtf(var + BN_EPS) * gamma[c] + beta[c];
    Y[i] = v > 0.0f ? v : 0.0f;
}

// ---------------------------------------------------------------------------
__global__ void maxpool_kernel(const float* __restrict__ Y, float* __restrict__ feat,
                               int BS, int ns, int C, int Ctot, int coff) {
    int i = blockIdx.x * blockDim.x + threadIdx.x;
    if (i >= BS * C) return;
    const int p = i / C, c = i % C;
    float m = -1e30f;
    for (int k = 0; k < ns; ++k) m = fmaxf(m, Y[(size_t)(p * ns + k) * C + c]);
    feat[(size_t)p * Ctot + coff + c] = m;
}

// ---------------------------------------------------------------------------
__global__ void knn_interp_kernel(const float* __restrict__ to_xyz, const float* __restrict__ from_xyz,
                                  const float* __restrict__ from_feat, const float* __restrict__ to_feat,
                                  float* __restrict__ out, int Np, int S, int Cf, int Ct,
                                  int total /* B*Np */) {
    int i = blockIdx.x * blockDim.x + threadIdx.x;
    if (i >= total) return;
    const int b = i / Np;
    const float px = to_xyz[i * 3 + 0];
    const float py = to_xyz[i * 3 + 1];
    const float pz = to_xyz[i * 3 + 2];
    float d0 = 1e30f, d1 = 1e30f, d2 = 1e30f;
    int   i0 = 0, i1 = 0, i2 = 0;
    for (int s = 0; s < S; ++s) {
        const float dx = from_xyz[(b * S + s) * 3 + 0] - px;
        const float dy = from_xyz[(b * S + s) * 3 + 1] - py;
        const float dz = from_xyz[(b * S + s) * 3 + 2] - pz;
        const float d  = dx * dx + dy * dy + dz * dz;
        if (d < d0)      { d2 = d1; i2 = i1; d1 = d0; i1 = i0; d0 = d; i0 = s; }
        else if (d < d1) { d2 = d1; i2 = i1; d1 = d;  i1 = s; }
        else if (d < d2) { d2 = d;  i2 = s; }
    }
    float w0 = 1.0f / (d0 + 1e-8f);
    float w1 = 1.0f / (d1 + 1e-8f);
    float w2 = 1.0f / (d2 + 1e-8f);
    const float ws = w0 + w1 + w2;
    w0 /= ws; w1 /= ws; w2 /= ws;
    float* o = out + (size_t)i * (Ct + Cf);
    if (Ct > 0) {
        const float* tf = to_feat + (size_t)i * Ct;
        for (int c = 0; c < Ct; ++c) o[c] = tf[c];
    }
    const float* f = from_feat + (size_t)b * S * Cf;
    for (int c = 0; c < Cf; ++c)
        o[Ct + c] = w0 * f[(size_t)i0 * Cf + c] + w1 * f[(size_t)i1 * Cf + c] + w2 * f[(size_t)i2 * Cf + c];
}

// ---------------------------------------------------------------------------
__global__ void conv2_logsoftmax_kernel(const float* __restrict__ h, const float* __restrict__ w,
                                        const float* __restrict__ bias, float* __restrict__ out,
                                        int total /* B*N */, int C, int K) {
    int i = blockIdx.x * blockDim.x + threadIdx.x;
    if (i >= total) return;
    float logits[16];
    for (int k = 0; k < K; ++k) {
        float s = bias[k];
        const float* hp = h + (size_t)i * C;
        for (int c = 0; c < C; ++c) s += hp[c] * w[c * K + k];
        logits[k] = s;
    }
    float mx = -1e30f;
    for (int k = 0; k < K; ++k) mx = fmaxf(mx, logits[k]);
    float sum = 0.0f;
    for (int k = 0; k < K; ++k) sum += expf(logits[k] - mx);
    const float lse = mx + logf(sum);
    for (int k = 0; k < K; ++k) out[(size_t)i * K + k] = logits[k] - lse;
}

// ---------------------------------------------------------------------------
extern "C" void kernel_launch(void* const* d_in, const int* in_sizes, int n_in,
                              void* d_out, int out_size, void* d_ws, size_t ws_size,
                              hipStream_t stream) {
    const int B  = in_sizes[0] / (4096 * 9);
    const int N0 = 4096;
    const float* x = (const float*)d_in[0];

    // ---- parameter unpack (dict order: sa1..sa4, fp1..fp4, head1, conv2) ----
    struct LP { const float *w, *b, *g, *be; };
    int cur = 1;
    auto nextLP = [&]() { LP L; L.w = (const float*)d_in[cur++]; L.b = (const float*)d_in[cur++];
                          L.g = (const float*)d_in[cur++]; L.be = (const float*)d_in[cur++]; return L; };
    LP saL[4][2][3];
    for (int st = 0; st < 4; ++st)
        for (int br = 0; br < 2; ++br)
            for (int l = 0; l < 3; ++l) saL[st][br][l] = nextLP();
    LP fp1L[2], fp2L[2], fp3L[2], fp4L[3], headL;
    for (int l = 0; l < 2; ++l) fp1L[l] = nextLP();
    for (int l = 0; l < 2; ++l) fp2L[l] = nextLP();
    for (int l = 0; l < 2; ++l) fp3L[l] = nextLP();
    for (int l = 0; l < 3; ++l) fp4L[l] = nextLP();
    headL = nextLP();
    const float* c2w = (const float*)d_in[cur++];
    const float* c2b = (const float*)d_in[cur++];

    // ---- workspace carve ----
    char* wsb = (char*)d_ws;
    size_t off = 0;
    auto alloc = [&](size_t bytes) { void* p = wsb + off; off = (off + bytes + 255) & ~(size_t)255; return p; };
    float* xyz0   = (float*)alloc((size_t)B * N0 * 3 * 4);
    float* dist   = (float*)alloc((size_t)B * N0 * 4);
    int*   fpsidx = (int*)  alloc((size_t)B * 1024 * 4);
    int*   gidx   = (int*)  alloc((size_t)B * 1024 * 32 * 4);
    float* lx[4], *lf[4];
    const int   Spts[4] = {1024, 256, 64, 16};
    const int   Ctot[4] = {96, 256, 512, 1024};
    for (int s = 0; s < 4; ++s) lx[s] = (float*)alloc((size_t)B * Spts[s] * 3 * 4);
    for (int s = 0; s < 4; ++s) lf[s] = (float*)alloc((size_t)B * Spts[s] * Ctot[s] * 4);
    float* f1   = (float*)alloc((size_t)B * 64 * 256 * 4);
    float* f2   = (float*)alloc((size_t)B * 256 * 256 * 4);
    float* f3   = (float*)alloc((size_t)B * 1024 * 128 * 4);
    float* f4   = (float*)alloc((size_t)B * 4096 * 128 * 4);
    float* hbuf = (float*)alloc((size_t)B * 4096 * 128 * 4);
    float* stats = (float*)alloc(2 * 512 * 4);
    const size_t BUFN = (size_t)B * 1024 * 32 * 64;            // max activation: 262144 x 64 f32
    float* bufA = (float*)alloc(BUFN * 4);
    float* bufB = (float*)alloc(BUFN * 4);
    _Float16* Xp = (_Float16*)alloc((size_t)B * 1024 * 32 * 32 * 2);  // max M*Kp halves (262144*32)
    _Float16* Wp = (_Float16*)alloc((size_t)1536 * 256 * 2);          // max Kp*Np halves

    // ---- GEMM + BN + ReLU helper ----
    auto gemm_bn = [&](const float* X, float* Y, int M, int K, int Nc, const LP& L) {
        const int nkt = (K + 31) / 32;
        const int nct = (Nc + 15) / 16;
        const int chA = (M / 16) * nkt * 64;                   // 16-B chunks in Xp
        const int chB = nct * nkt * 64;
        pack_a_kernel<<<(chA + 255) / 256, 256, 0, stream>>>(X, Xp, K, nkt, chA);
        pack_b_kernel<<<(chB + 255) / 256, 256, 0, stream>>>(L.w, Wp, K, Nc, nkt, chB);
        dim3 grid(nct, M / 64);                                // M % 64 == 0 by construction
        gemm_packed_wmma<<<grid, 128, 0, stream>>>(Xp, Wp, L.b, Y, M, Nc, nkt);
        hipMemsetAsync(stats, 0, 2 * Nc * sizeof(float), stream);
        dim3 g2((Nc + 63) / 64, (M + 4095) / 4096);
        bn_stats_kernel<<<g2, 64, 0, stream>>>(Y, stats, M, Nc);
        long long tot = (long long)M * Nc;
        bn_relu_kernel<<<(unsigned)((tot + 255) / 256), 256, 0, stream>>>(Y, stats, L.g, L.be, M, Nc, tot);
    };

    extract_xyz_kernel<<<(B * N0 + 255) / 256, 256, 0, stream>>>(x, xyz0, B * N0);

    // ---- set abstraction stages ----
    static const int   sadims[4][2][3] = {{{16,16,32},{32,32,64}},
                                          {{64,64,128},{64,96,128}},
                                          {{128,196,256},{128,196,256}},
                                          {{256,256,512},{256,384,512}}};
    static const float sarad[4][2]     = {{0.05f,0.1f},{0.1f,0.2f},{0.2f,0.4f},{0.4f,0.8f}};
    static const int   sans[4][2]      = {{16,32},{16,32},{16,32},{16,32}};
    const int featC[4] = {9, 96, 256, 512};

    const float* xyz_in  = xyz0;
    const float* feat_in = x;
    int Nin = N0;
    for (int st = 0; st < 4; ++st) {
        const int S = Spts[st];
        fps_kernel<<<B, 256, 0, stream>>>(xyz_in, dist, fpsidx, Nin, S);
        gather_xyz_kernel<<<(B * S + 255) / 256, 256, 0, stream>>>(xyz_in, fpsidx, lx[st], Nin, B * S, S);
        int coff = 0;
        for (int br = 0; br < 2; ++br) {
            const int ns = sans[st][br];
            const int M  = B * S * ns;
            const float r = sarad[st][br];
            ball_group_kernel<<<(B * S + 255) / 256, 256, 0, stream>>>(xyz_in, lx[st], gidx, Nin, S, B * S, r * r, ns);
            group_concat_kernel<<<(M + 255) / 256, 256, 0, stream>>>(xyz_in, feat_in, lx[st], gidx, bufA,
                                                                     Nin, S, ns, featC[st], M);
            int cin = featC[st] + 3;
            float* in = bufA; float* out = bufB;
            for (int l = 0; l < 3; ++l) {
                gemm_bn(in, out, M, cin, sadims[st][br][l], saL[st][br][l]);
                cin = sadims[st][br][l];
                float* tmp = in; in = out; out = tmp;
            }
            maxpool_kernel<<<(B * S * cin + 255) / 256, 256, 0, stream>>>(in, lf[st], B * S, ns, cin, Ctot[st], coff);
            coff += cin;
        }
        xyz_in = lx[st]; feat_in = lf[st]; Nin = S;
    }

    // ---- feature propagation ----
    auto run_fp = [&](const float* to_xyz, int Np, const float* from_xyz, int S,
                      const float* to_feat, int Ct, const float* from_feat, int Cf,
                      const LP* layers, const int* dims, int nl, float* outFeat) {
        const int M = B * Np;
        knn_interp_kernel<<<(M + 255) / 256, 256, 0, stream>>>(to_xyz, from_xyz, from_feat, to_feat,
                                                               bufA, Np, S, Cf, Ct, M);
        int cin = Ct + Cf;
        float* in = bufA; float* out = bufB;
        for (int l = 0; l < nl; ++l) {
            gemm_bn(in, out, M, cin, dims[l], layers[l]);
            cin = dims[l];
            float* tmp = in; in = out; out = tmp;
        }
        hipMemcpyAsync(outFeat, in, (size_t)M * cin * sizeof(float), hipMemcpyDeviceToDevice, stream);
    };

    static const int fp1d[2] = {256, 256};
    static const int fp2d[2] = {256, 256};
    static const int fp3d[2] = {256, 128};
    static const int fp4d[3] = {128, 128, 128};
    run_fp(lx[2], 64,   lx[3], 16,  lf[2], 512, lf[3], 1024, fp1L, fp1d, 2, f1);
    run_fp(lx[1], 256,  lx[2], 64,  lf[1], 256, f1,    256,  fp2L, fp2d, 2, f2);
    run_fp(lx[0], 1024, lx[1], 256, lf[0], 96,  f2,    256,  fp3L, fp3d, 2, f3);
    run_fp(xyz0,  4096, lx[0], 1024, nullptr, 0, f3,   128,  fp4L, fp4d, 3, f4);

    // ---- head + final projection + log_softmax ----
    gemm_bn(f4, hbuf, B * 4096, 128, 128, headL);
    conv2_logsoftmax_kernel<<<(B * 4096 + 255) / 256, 256, 0, stream>>>(hbuf, c2w, c2b, (float*)d_out,
                                                                        B * 4096, 128, 13);
}
